// MoE_21723944583386
// MI455X (gfx1250) — matrix operand order repaired and checked by reading, not compiled, not run on hardware
//
#include <hip/hip_runtime.h>

#define T   4096
#define Dm  1024
#define Hm  2048
#define Em  8
#define Km  2
#define NOISE_EPS 0.01f
#define CVLOSS 0.01f
#define SWITCHLOSS 0.1f
#define ZLOSS 0.0001f

typedef __attribute__((ext_vector_type(16))) __bf16 v16bf;
typedef __attribute__((ext_vector_type(8)))  float  v8f;

union AFrag { uint4 q[2]; v16bf v; };

// ---------------- workspace layout (bytes) ----------------
// [0,32)      int   cnt[8]
// [32,64)     float probs_sum[8]
// [64,96)     float gates_sum[8]
// [96,100)    float zl_sum
// [128, 128+E*T*4)          int   list[E][T]
// [128+E*T*4, 128+2*E*T*4)  float glist[E][T]

__global__ void moe_zero_kernel(float* __restrict__ y, float* __restrict__ wshdr) {
  size_t i = (size_t)blockIdx.x * blockDim.x + threadIdx.x;
  size_t stride = (size_t)gridDim.x * blockDim.x;
  for (size_t k = i; k < (size_t)T * Dm; k += stride) y[k] = 0.0f;
  if (i < 32) wshdr[i] = 0.0f;  // zero bits also valid for the int counters
}

__device__ __forceinline__ float softplus_f(float x) {
  return x > 20.0f ? x : log1pf(__expf(x));
}

__global__ void moe_gate_kernel(const float* __restrict__ x,
                                const float* __restrict__ wg,
                                const float* __restrict__ eps,
                                int* __restrict__ cnt,
                                float* __restrict__ probs_sum,
                                float* __restrict__ gates_sum,
                                float* __restrict__ zl_sum,
                                int* __restrict__ list,
                                float* __restrict__ glist) {
  __shared__ float dots[16][16];
  const int tid = threadIdx.x;
  const int col = tid & 15;      // 0..15 output column of w_gate
  const int tk  = tid >> 4;      // 0..15 token within block
  const int t   = blockIdx.x * 16 + tk;
  // dot: x[t,:] @ w_gate[:,col]
  float acc = 0.0f;
  const float* xr = x + (size_t)t * Dm;
  #pragma unroll 4
  for (int d = 0; d < Dm; ++d) acc += xr[d] * wg[(size_t)d * 16 + col];
  dots[tk][col] = acc;
  __syncthreads();

  if (tid < 16) {
    const int tt = blockIdx.x * 16 + tid;
    float l[8];
    float mx = -3.4e38f;
    #pragma unroll
    for (int e = 0; e < Em; ++e) {
      float stddev = softplus_f(dots[tid][8 + e]) + NOISE_EPS;
      l[e] = dots[tid][e] + eps[(size_t)tt * Em + e] * stddev;
      mx = fmaxf(mx, l[e]);
    }
    float p[8]; float s = 0.0f;
    #pragma unroll
    for (int e = 0; e < Em; ++e) { p[e] = __expf(l[e] - mx); s += p[e]; }
    float inv = 1.0f / s;
    #pragma unroll
    for (int e = 0; e < Em; ++e) p[e] *= inv;
    float lse = mx + logf(s);
    // top-2 (first index wins ties, matching top_k)
    int i1 = 0; float g1 = p[0];
    #pragma unroll
    for (int e = 1; e < Em; ++e) if (p[e] > g1) { g1 = p[e]; i1 = e; }
    int i2 = -1; float g2 = -1.0f;
    #pragma unroll
    for (int e = 0; e < Em; ++e) if (e != i1 && p[e] > g2) { g2 = p[e]; i2 = e; }
    // loss accumulators
    #pragma unroll
    for (int e = 0; e < Em; ++e) atomicAdd(&probs_sum[e], p[e]);
    atomicAdd(&gates_sum[i1], g1);
    atomicAdd(&gates_sum[i2], g2);
    atomicAdd(zl_sum, lse * lse);
    // routing lists
    int p1 = atomicAdd(&cnt[i1], 1);
    list[i1 * T + p1] = tt;  glist[i1 * T + p1] = g1;
    int p2 = atomicAdd(&cnt[i2], 1);
    list[i2 * T + p2] = tt;  glist[i2 * T + p2] = g2;
  }
}

__global__ void moe_loss_kernel(const int* __restrict__ cnt,
                                const float* __restrict__ probs_sum,
                                const float* __restrict__ gates_sum,
                                const float* __restrict__ zl_sum,
                                float* __restrict__ out_loss) {
  float psum = 0.0f, csum = 0.0f, gsum = 0.0f;
  for (int e = 0; e < Em; ++e) {
    psum += probs_sum[e];
    csum += (float)cnt[e];
    gsum += gates_sum[e];
  }
  float dot = 0.0f;
  for (int e = 0; e < Em; ++e)
    dot += (probs_sum[e] / psum) * ((float)cnt[e] / csum);
  float sw = (1.0f - dot) * (float)Em;
  float mean = 0.0f;
  float gn[8];
  for (int e = 0; e < Em; ++e) { gn[e] = gates_sum[e] / gsum; mean += gn[e]; }
  mean *= (1.0f / Em);
  float var = 0.0f;
  for (int e = 0; e < Em; ++e) { float d = gn[e] - mean; var += d * d; }
  var *= (1.0f / (Em - 1));           // unbiased
  float cv = var / (mean * mean + 1e-10f);
  float zl = zl_sum[0] / (float)T;
  *out_loss = CVLOSS * cv + SWITCHLOSS * sw + ZLOSS * zl;
}

// ---------------- fused expert FFN: relu(X@W1)*gate @ W2, atomically into y ----
// block = (expert, 16-token tile), 256 threads = 8 waves (wave32)
__global__ void __launch_bounds__(256)
moe_ffn_kernel(const float* __restrict__ x,
               const float* __restrict__ w1,
               const float* __restrict__ w2,
               float* __restrict__ y,
               const int* __restrict__ cnt,
               const int* __restrict__ list,
               const float* __restrict__ glist) {
  const int e    = blockIdx.x >> 8;     // 256 tiles max per expert
  const int tile = blockIdx.x & 255;
  const int ce = cnt[e];
  if (tile * 16 >= ce) return;

  __shared__ __bf16 Xs[16][Dm + 8];     // +8 keeps 16B-aligned rows
  __shared__ __bf16 Hs[16][Hm + 8];
  __shared__ int   stok[16];
  __shared__ float sgate[16];

  const int tid = threadIdx.x;
  if (tid < 16) {
    int i = tile * 16 + tid;
    if (i < ce) { stok[tid] = list[e * T + i]; sgate[tid] = glist[e * T + i]; }
    else        { stok[tid] = 0;              sgate[tid] = 0.0f; }
  }
  __syncthreads();

  // gather + f32->bf16 convert the 16 token rows into LDS
  for (int idx = tid; idx < 16 * Dm; idx += 256) {
    int r = idx >> 10, c = idx & (Dm - 1);
    Xs[r][c] = (__bf16)x[(size_t)stok[r] * Dm + c];
  }
  __syncthreads();

  const int wv   = tid >> 5;
  const int lane = tid & 31;
  const int m    = lane & 15;           // A row / B col / C col within tile
  const int half = lane >> 4;           // K-half selector
  const int koffA = half * 16;

  // hoist per-lane gates and output-row bases into registers (once, not per tile)
  float  gr[8];
  size_t ybase[8];
  #pragma unroll
  for (int v = 0; v < 8; ++v) {
    int row = v + half * 8;
    gr[v]    = sgate[row];
    ybase[v] = (size_t)stok[row] * Dm;
  }

  // ---- GEMM1: H[16,2048] = relu(X[16,1024] @ W1[e]) * gate, bf16 into LDS ----
  for (int nt = 0; nt < 16; ++nt) {
    const int nb = wv * 256 + nt * 16;
    const int n  = nb + m;
    v8f acc = {};
    #pragma unroll 2
    for (int kb = 0; kb < Dm; kb += 32) {
      AFrag a;
      a.q[0] = *(const uint4*)&Xs[m][kb + koffA];
      a.q[1] = *(const uint4*)&Xs[m][kb + koffA + 8];
      const int kc = kb + half * 16;
      const float* wp = w1 + ((size_t)e * Dm + kc) * Hm + n;
      __builtin_prefetch(wp + (size_t)32 * Hm, 0, 1);   // next K-slab
      v16bf b;
      #pragma unroll
      for (int j = 0; j < 16; ++j) b[j] = (__bf16)wp[(size_t)j * Hm];
      acc = __builtin_amdgcn_wmma_f32_16x16x32_bf16(
          false, a.v, false, b, (short)0, acc, false, false);
    }
    // branchless relu * gate, bf16 store
    #pragma unroll
    for (int v = 0; v < 8; ++v) {
      float hv = fmaxf(acc[v], 0.0f) * gr[v];
      Hs[v + half * 8][nb + m] = (__bf16)hv;
    }
  }
  __syncthreads();

  // ---- GEMM2: Y[16,1024] = H[16,2048] @ W2[e], atomic f32 add into y ----
  for (int nt = 0; nt < 8; ++nt) {
    const int nb = wv * 128 + nt * 16;
    const int n  = nb + m;
    v8f acc = {};
    #pragma unroll 2
    for (int kb = 0; kb < Hm; kb += 32) {
      AFrag a;
      a.q[0] = *(const uint4*)&Hs[m][kb + koffA];
      a.q[1] = *(const uint4*)&Hs[m][kb + koffA + 8];
      const int kc = kb + half * 16;
      const float* wp = w2 + ((size_t)e * Hm + kc) * Dm + n;
      __builtin_prefetch(wp + (size_t)32 * Dm, 0, 1);   // next K-slab
      v16bf b;
      #pragma unroll
      for (int j = 0; j < 16; ++j) b[j] = (__bf16)wp[(size_t)j * Dm];
      acc = __builtin_amdgcn_wmma_f32_16x16x32_bf16(
          false, a.v, false, b, (short)0, acc, false, false);
    }
    #pragma unroll
    for (int v = 0; v < 8; ++v) {
      __hip_atomic_fetch_add(&y[ybase[v] + nb + m], acc[v],
                             __ATOMIC_RELAXED, __HIP_MEMORY_SCOPE_AGENT);
    }
  }
}

extern "C" void kernel_launch(void* const* d_in, const int* in_sizes, int n_in,
                              void* d_out, int out_size, void* d_ws, size_t ws_size,
                              hipStream_t stream) {
  const float* x   = (const float*)d_in[0];
  const float* wg  = (const float*)d_in[1];
  const float* w1  = (const float*)d_in[2];
  const float* w2  = (const float*)d_in[3];
  const float* eps = (const float*)d_in[4];
  float* out = (float*)d_out;              // [T*D] y, then [1] loss

  char* ws = (char*)d_ws;
  int*   cnt       = (int*)(ws + 0);
  float* probs_sum = (float*)(ws + 32);
  float* gates_sum = (float*)(ws + 64);
  float* zl_sum    = (float*)(ws + 96);
  int*   list      = (int*)(ws + 128);
  float* glist     = (float*)(ws + 128 + (size_t)Em * T * 4);

  moe_zero_kernel<<<2048, 256, 0, stream>>>(out, (float*)ws);
  moe_gate_kernel<<<T / 16, 256, 0, stream>>>(x, wg, eps, cnt, probs_sum,
                                              gates_sum, zl_sum, list, glist);
  moe_loss_kernel<<<1, 1, 0, stream>>>(cnt, probs_sum, gates_sum, zl_sum,
                                       out + (size_t)T * Dm);
  moe_ffn_kernel<<<Em * 256, 256, 0, stream>>>(x, w1, w2, out, cnt, list, glist);
}
